// Xl_72052371357920
// MI455X (gfx1250) — matrix operand order repaired and checked by reading, not compiled, not run on hardware
//
#include <hip/hip_runtime.h>
#include <hip/hip_bf16.h>

typedef unsigned short u16;
typedef __attribute__((ext_vector_type(16))) __bf16 bf16x16;
typedef __attribute__((ext_vector_type(8)))  __bf16 bf16x8;
typedef __attribute__((ext_vector_type(8)))  float  f32x8;
typedef int v4i_ __attribute__((vector_size(16)));   // matches builtin param type

#define FLAG_OUT_BF16 1
#define FLAG_BIAS     2
#define FLAG_RELU     4

// ---------------------------------------------------------------- helpers
__device__ inline u16 f2bf(float f) {
  unsigned int u = __builtin_bit_cast(unsigned int, f);
  u += 0x7FFFu + ((u >> 16) & 1u);           // round-to-nearest-even
  return (u16)(u >> 16);
}
__device__ inline float bf2f(u16 h) {
  unsigned int u = ((unsigned int)h) << 16;
  return __builtin_bit_cast(float, u);
}
__device__ inline bf16x8 ld8(const u16* p) {
  return __builtin_bit_cast(bf16x8, *(const uint4*)p);
}
__device__ inline bf16x16 combine16(bf16x8 lo, bf16x8 hi) {
  bf16x16 r;
#pragma unroll
  for (int i = 0; i < 8; ++i) { r[i] = lo[i]; r[i + 8] = hi[i]; }
  return r;
}
// A-matrix fragment (16x32 bf16). rowp = this lane's row (m = lane&15).
// Lane-half selects 16 contiguous k values {half*16..+16} (matches ld_bfrag).
__device__ inline bf16x16 ld_afrag(const u16* rowp, int half) {
  return combine16(ld8(rowp + half * 16), ld8(rowp + half * 16 + 8));
}
// B-matrix fragment (32x16 bf16): 16 contiguous k values (caller adds half*16).
__device__ inline bf16x16 ld_bfrag(const u16* p) {
  return combine16(ld8(p), ld8(p + 8));
}
__device__ inline f32x8 wmma_bf16(bf16x16 a, bf16x16 b, f32x8 c) {
  return __builtin_amdgcn_wmma_f32_16x16x32_bf16(false, a, false, b, (short)0, c,
                                                 false, false);
}

// ---------------- async global->LDS staging (CDNA5 GLOBAL_LOAD_ASYNC_TO_LDS)
#if defined(__HIP_DEVICE_COMPILE__) && \
    __has_builtin(__builtin_amdgcn_global_load_async_to_lds_b128)
#define HAVE_ASYNC_LDS 1
#else
#define HAVE_ASYNC_LDS 0
#endif

__device__ inline void async_ld_b128(const void* g, void* l) {
#if HAVE_ASYNC_LDS
  // Builtin signature (from clang diagnostic): (AS1 int4*, AS3 int4*, imm, imm).
  // LDS offset = low 32 bits of the generic address (flat aperture rule).
  __builtin_amdgcn_global_load_async_to_lds_b128(
      (__attribute__((address_space(1))) v4i_*)(unsigned long long)g,
      (__attribute__((address_space(3))) v4i_*)(unsigned long long)(unsigned)(unsigned long long)l,
      0, 0);
#else
  *(uint4*)l = *(const uint4*)g;
#endif
}
__device__ inline void async_wait0() {
#if HAVE_ASYNC_LDS
#if __has_builtin(__builtin_amdgcn_s_wait_asynccnt)
  __builtin_amdgcn_s_wait_asynccnt(0);
#else
  asm volatile("s_wait_asynccnt 0" ::: "memory");
#endif
#endif
}

// ---------------------------------------------------------------- GEMM
// C[M,N] = A[M,K]bf16 @ B[K,N]bf16 (+bias)(+relu), out f32 or bf16.
// Block tile 128x128, K-step 32, 256 threads = 8 waves (2Mx4N), wave tile 64x32.
// Double-buffered LDS: A staged via async global->LDS copies (ASYNCcnt),
// B software-pipelined through registers into a transposed LDS tile.
template <int FLAGS>
__global__ __launch_bounds__(256)
void gemm_bf16_kernel(const u16* __restrict__ A, const u16* __restrict__ B,
                      const float* __restrict__ bias, float* __restrict__ Cf,
                      u16* __restrict__ Cb, int M, int N, int Kd) {
  __shared__ __align__(16) u16 As[2][128 * 32];
  __shared__ __align__(16) u16 Bs[2][128 * 32];   // transposed: [n][k]
  const int tid = threadIdx.x;
  const int bm = blockIdx.y * 128, bn = blockIdx.x * 128;
  const int wave = tid >> 5, lane = tid & 31;
  const int l16 = lane & 15, half = lane >> 4;
  const int wm = (wave >> 2) * 64;
  const int wn = (wave & 3) * 32;
  const int ar = tid >> 1, ac = (tid & 1) * 16;     // A staging: 2 thr/row
  const int bk = tid & 31, bn16 = (tid >> 5) * 16;  // B staging: 16 n per thr

  uint4 breg0, breg1;

  // prologue: stage tile 0
  {
    const u16* src = A + (size_t)(bm + ar) * Kd + ac;
    u16* dst = &As[0][ar * 32 + ac];
    async_ld_b128(src, dst);
    async_ld_b128(src + 8, dst + 8);
    const u16* bsrc = B + (size_t)bk * N + bn + bn16;
    breg0 = *(const uint4*)bsrc;
    breg1 = *(const uint4*)(bsrc + 8);
    u16 tmp[16];
    *(uint4*)tmp = breg0; *(uint4*)(tmp + 8) = breg1;
#pragma unroll
    for (int i = 0; i < 16; ++i) Bs[0][(bn16 + i) * 32 + bk] = tmp[i];
  }

  f32x8 acc[4][2] = {};
  const int nk = Kd >> 5;
  for (int i = 0; i < nk; ++i) {
    const int cur = i & 1, nxt = cur ^ 1;
    // pipeline next B tile into registers before the barrier
    if (i + 1 < nk) {
      const u16* bsrc = B + (size_t)((i + 1) * 32 + bk) * N + bn + bn16;
      breg0 = *(const uint4*)bsrc;
      breg1 = *(const uint4*)(bsrc + 8);
    }
    async_wait0();        // this wave's async A copies into buf `cur` done
    __syncthreads();      // cur tiles visible block-wide; nxt free to overwrite
    if (i + 1 < nk) {
      const u16* src = A + (size_t)(bm + ar) * Kd + (i + 1) * 32 + ac;
      u16* dst = &As[nxt][ar * 32 + ac];
      async_ld_b128(src, dst);
      async_ld_b128(src + 8, dst + 8);
      u16 tmp[16];
      *(uint4*)tmp = breg0; *(uint4*)(tmp + 8) = breg1;
#pragma unroll
      for (int j = 0; j < 16; ++j) Bs[nxt][(bn16 + j) * 32 + bk] = tmp[j];
    }
    // compute on buf `cur`
    bf16x16 bfrag[2];
#pragma unroll
    for (int nf = 0; nf < 2; ++nf)
      bfrag[nf] = ld_bfrag(&Bs[cur][(wn + nf * 16 + l16) * 32 + half * 16]);
#pragma unroll
    for (int mf = 0; mf < 4; ++mf) {
      bf16x16 afrag = ld_afrag(&As[cur][(wm + mf * 16 + l16) * 32], half);
#pragma unroll
      for (int nf = 0; nf < 2; ++nf)
        acc[mf][nf] = wmma_bf16(afrag, bfrag[nf], acc[mf][nf]);
    }
  }
  // epilogue (compile-time flags: no branches, one bias load per column tile)
#pragma unroll
  for (int nf = 0; nf < 2; ++nf) {
    const int col = bn + wn + nf * 16 + l16;
    float bv = 0.0f;
    if (FLAGS & FLAG_BIAS) bv = bias[col];
#pragma unroll
    for (int mf = 0; mf < 4; ++mf)
#pragma unroll
      for (int v = 0; v < 8; ++v) {
        int row = bm + wm + mf * 16 + half * 8 + v;  // C layout: m = vgpr+half*8
        float x = acc[mf][nf][v] + bv;
        if (FLAGS & FLAG_RELU) x = fmaxf(x, 0.0f);
        if (FLAGS & FLAG_OUT_BF16) Cb[(size_t)row * N + col] = f2bf(x);
        else                       Cf[(size_t)row * N + col] = x;
      }
  }
}

// ---------------------------------------------------------------- attention
// Grid (Q/64, H, B), 128 threads = 4 waves; each wave owns 16 q-rows.
// Rel-shift folded: BD_shift[q,k] = BDraw[q, k + Q-1-q], computed as a banded
// 16x80 WMMA window per 64-key tile, gathered through LDS.
#define QL 1024
#define MLEN 1024
#define KLEN 2048
__global__ __launch_bounds__(128)
void attn_kernel(const u16* __restrict__ heads,   // [B, KLEN, 3072] q|k|v
                 const u16* __restrict__ qrw,     // [B, QL, 1024]  q + r_w_bias
                 const u16* __restrict__ qrr,     // [B, QL, 1024]  q + r_r_bias
                 const u16* __restrict__ rbuf,    // [KLEN, 1024]   r = p@W_r
                 u16* __restrict__ attnout) {     // [B, QL, 1024]
  const int qb = blockIdx.x * 64;
  const int h  = blockIdx.y;
  const int b  = blockIdx.z;
  const int tid = threadIdx.x, wave = tid >> 5, lane = tid & 31;
  const int l16 = lane & 15, half = lane >> 4;
  const int qw = qb + wave * 16;

  __shared__ __align__(16) u16  Vt[64][64];        // [dh][k] transposed V tile
  __shared__            float   BDw[4][16][80];    // per-wave BD window
  __shared__ __align__(16) u16  Pw[4][16][64];     // per-wave probs (bf16)

  // Persistent A-fragments: rows m = l16 of (q+rwb) and (q+rrb), dh 0..31/32..63
  bf16x16 aw[2], arr[2];
  {
    const u16* qp = qrw + ((size_t)b * QL + qw + l16) * 1024 + h * 64;
    aw[0] = ld_afrag(qp, half);
    aw[1] = ld_afrag(qp + 32, half);
    const u16* rp = qrr + ((size_t)b * QL + qw + l16) * 1024 + h * 64;
    arr[0] = ld_afrag(rp, half);
    arr[1] = ld_afrag(rp + 32, half);
  }
  f32x8 accO[4] = {};
  float rm[8], rl[8];
#pragma unroll
  for (int v = 0; v < 8; ++v) { rm[v] = -3.0e38f; rl[v] = 0.0f; }

  const int kend = qb + MLEN + 64;    // last key any row in this block attends to
  for (int k0 = 0; k0 < kend; k0 += 64) {
    __syncthreads();                  // protect Vt from previous iteration
    {   // stage V tile transposed: thread t -> k = t&63, dh-half = (t>>6)*32
      int kk = tid & 63, dh0 = (tid >> 6) * 32;
      const u16* vp = heads + ((size_t)b * KLEN + k0 + kk) * 3072 + 2048 + h * 64 + dh0;
      u16 tmp[32];
      *(uint4*)(tmp)      = *(const uint4*)(vp);
      *(uint4*)(tmp + 8)  = *(const uint4*)(vp + 8);
      *(uint4*)(tmp + 16) = *(const uint4*)(vp + 16);
      *(uint4*)(tmp + 24) = *(const uint4*)(vp + 24);
#pragma unroll
      for (int i = 0; i < 32; ++i) Vt[dh0 + i][kk] = tmp[i];
    }
    // BD banded window: j = k + Q-1 - q, window base jb covers all (q,k) of tile
    const int jb = k0 + QL - 16 - qw;
#pragma unroll
    for (int jt = 0; jt < 5; ++jt) {
      int j = jb + jt * 16 + l16;
      j = min(max(j, 0), KLEN - 1);   // out-of-range j only feeds masked scores
      const u16* rp = rbuf + (size_t)j * 1024 + h * 64 + half * 16;
      bf16x16 rb0 = ld_bfrag(rp);
      bf16x16 rb1 = ld_bfrag(rp + 32);
      f32x8 bd = {};
      bd = wmma_bf16(arr[0], rb0, bd);
      bd = wmma_bf16(arr[1], rb1, bd);
#pragma unroll
      for (int v = 0; v < 8; ++v)
        BDw[wave][v + half * 8][jt * 16 + l16] = bd[v];
    }
    __syncthreads();                  // Vt + BDw visible

    // AC tiles + combine + mask
    float sc[4][8];
#pragma unroll
    for (int nt = 0; nt < 4; ++nt) {
      int key = k0 + nt * 16 + l16;
      const u16* kp = heads + ((size_t)b * KLEN + key) * 3072 + 1024 + h * 64 + half * 16;
      bf16x16 kb0 = ld_bfrag(kp);
      bf16x16 kb1 = ld_bfrag(kp + 32);
      f32x8 ac = {};
      ac = wmma_bf16(aw[0], kb0, ac);
      ac = wmma_bf16(aw[1], kb1, ac);
#pragma unroll
      for (int v = 0; v < 8; ++v) {
        int m = v + half * 8;
        float bdv = BDw[wave][m][nt * 16 + l16 + 15 - m];   // diagonal gather
        float s = (ac[v] + bdv) * 0.125f;                   // 1/sqrt(DH)
        if (key > qw + m + MLEN) s = -1.0e30f;              // causal+memory mask
        sc[nt][v] = s;
      }
    }
    // online softmax per row (columns live across the 16-lane half-groups)
#pragma unroll
    for (int v = 0; v < 8; ++v) {
      float mx = fmaxf(fmaxf(sc[0][v], sc[1][v]), fmaxf(sc[2][v], sc[3][v]));
#pragma unroll
      for (int off = 1; off < 16; off <<= 1) mx = fmaxf(mx, __shfl_xor(mx, off, 32));
      float mnew = fmaxf(rm[v], mx);
      float corr = __expf(rm[v] - mnew);
      rm[v] = mnew;
      float lsum = 0.0f;
#pragma unroll
      for (int nt = 0; nt < 4; ++nt) {
        float p = __expf(sc[nt][v] - mnew);
        sc[nt][v] = p;
        lsum += p;
      }
#pragma unroll
      for (int off = 1; off < 16; off <<= 1) lsum += __shfl_xor(lsum, off, 32);
      rl[v] = rl[v] * corr + lsum;
#pragma unroll
      for (int dt = 0; dt < 4; ++dt) accO[dt][v] *= corr;
    }
    // restage probs as bf16 A-fragments
#pragma unroll
    for (int nt = 0; nt < 4; ++nt)
#pragma unroll
      for (int v = 0; v < 8; ++v)
        Pw[wave][v + half * 8][nt * 16 + l16] = f2bf(sc[nt][v]);
    bf16x16 pa0 = ld_afrag(&Pw[wave][l16][0], half);
    bf16x16 pa1 = ld_afrag(&Pw[wave][l16][32], half);
#pragma unroll
    for (int dt = 0; dt < 4; ++dt) {
      bf16x16 vb0 = ld_bfrag(&Vt[dt * 16 + l16][half * 16]);
      bf16x16 vb1 = ld_bfrag(&Vt[dt * 16 + l16][32 + half * 16]);
      accO[dt] = wmma_bf16(pa0, vb0, accO[dt]);
      accO[dt] = wmma_bf16(pa1, vb1, accO[dt]);
    }
  }
#pragma unroll
  for (int dt = 0; dt < 4; ++dt)
#pragma unroll
    for (int v = 0; v < 8; ++v) {
      int m = v + half * 8;
      float o = accO[dt][v] / rl[v];
      attnout[((size_t)b * QL + qw + m) * 1024 + h * 64 + dt * 16 + l16] = f2bf(o);
    }
}

// ---------------------------------------------------------------- small kernels
__global__ __launch_bounds__(256)
void cvt_kernel(const float* __restrict__ s, u16* __restrict__ d) {
  int base = blockIdx.x * 1024 + threadIdx.x;
#pragma unroll
  for (int i = 0; i < 4; ++i) d[base + i * 256] = f2bf(s[base + i * 256]);
}

__global__ __launch_bounds__(256)
void pos_kernel(u16* __restrict__ p) {      // sinusoidal, rows = KLEN, D=1024
  int row = blockIdx.x;
  float pos = (float)(KLEN - 1 - row);
#pragma unroll
  for (int i = 0; i < 2; ++i) {
    int t = threadIdx.x + i * 256;          // 0..511
    float freq = __expf(-9.2103403719761836f * ((float)(2 * t) * (1.0f / 1024.0f)));
    float a = pos * freq;
    p[row * 1024 + t]       = f2bf(__sinf(a));
    p[row * 1024 + 512 + t] = f2bf(__cosf(a));
  }
}

__global__ __launch_bounds__(256)
void concat_kernel(const float* __restrict__ mem_l, const float* __restrict__ x,
                   u16* __restrict__ xmem) {     // [B, KLEN, 1024] bf16
  int idx = blockIdx.x * 256 + threadIdx.x;
  int d = idx & 1023;
  int k = (idx >> 10) & 2047;
  int b = idx >> 21;
  float v = (k < 1024) ? mem_l[((size_t)b * 1024 + k) * 1024 + d]
                       : x[((size_t)b * 1024 + (k - 1024)) * 1024 + d];
  xmem[idx] = f2bf(v);
}

__global__ __launch_bounds__(256)
void qbias_kernel(const u16* __restrict__ heads, const float* __restrict__ rwb,
                  const float* __restrict__ rrb, u16* __restrict__ qrw,
                  u16* __restrict__ qrr) {
  int idx = blockIdx.x * 256 + threadIdx.x;   // over B*Q*1024
  int c = idx & 1023;
  int q = (idx >> 10) & 1023;
  int b = idx >> 20;
  float f = bf2f(heads[((size_t)b * 2048 + 1024 + q) * 3072 + c]);
  qrw[idx] = f2bf(f + rwb[c]);
  qrr[idx] = f2bf(f + rrb[c]);
}

__global__ __launch_bounds__(256)
void initx_kernel(const float* __restrict__ xin, float* __restrict__ xcur,
                  u16* __restrict__ xbf) {
  int idx = blockIdx.x * 256 + threadIdx.x;
  float v = xin[idx];
  xcur[idx] = v;
  xbf[idx] = f2bf(v);
}

__global__ __launch_bounds__(256)
void add_ln_kernel(const float* __restrict__ gout, float* __restrict__ x,
                   u16* __restrict__ xbf, const float* __restrict__ g,
                   const float* __restrict__ beta) {
  int row = blockIdx.x;
  const float* gr = gout + (size_t)row * 1024;
  float* xr = x + (size_t)row * 1024;
  u16* xb = xbf + (size_t)row * 1024;
  __shared__ float sred[16];
  float v[4], s = 0.0f, s2 = 0.0f;
#pragma unroll
  for (int i = 0; i < 4; ++i) {
    int c = threadIdx.x + i * 256;
    float t = xr[c] + gr[c];
    v[i] = t; s += t; s2 += t * t;
  }
#pragma unroll
  for (int o = 1; o < 32; o <<= 1) { s += __shfl_xor(s, o, 32); s2 += __shfl_xor(s2, o, 32); }
  if ((threadIdx.x & 31) == 0) { sred[threadIdx.x >> 5] = s; sred[8 + (threadIdx.x >> 5)] = s2; }
  __syncthreads();
  s = 0.0f; s2 = 0.0f;
#pragma unroll
  for (int w = 0; w < 8; ++w) { s += sred[w]; s2 += sred[8 + w]; }
  float mu = s * (1.0f / 1024.0f);
  float var = s2 * (1.0f / 1024.0f) - mu * mu;
  float rstd = rsqrtf(var + 1e-5f);
#pragma unroll
  for (int i = 0; i < 4; ++i) {
    int c = threadIdx.x + i * 256;
    float t = (v[i] - mu) * rstd * g[c] + beta[c];
    xr[c] = t; xb[c] = f2bf(t);
  }
}

// ---------------------------------------------------------------- host
extern "C" void kernel_launch(void* const* d_in, const int* in_sizes, int n_in,
                              void* d_out, int out_size, void* d_ws, size_t ws_size,
                              hipStream_t stream) {
  (void)in_sizes; (void)n_in; (void)out_size; (void)ws_size;
  const float* x_in  = (const float*)d_in[0];
  const float* mem   = (const float*)d_in[1];
  const float* W_qkv = (const float*)d_in[2];
  const float* W_r   = (const float*)d_in[3];
  const float* W_o   = (const float*)d_in[4];
  const float* rwb   = (const float*)d_in[5];
  const float* rrb   = (const float*)d_in[6];
  const float* ln1g  = (const float*)d_in[7];
  const float* ln1b  = (const float*)d_in[8];
  const float* ln2g  = (const float*)d_in[9];
  const float* ln2b  = (const float*)d_in[10];
  const float* W_ff1 = (const float*)d_in[11];
  const float* b_ff1 = (const float*)d_in[12];
  const float* W_ff2 = (const float*)d_in[13];
  const float* b_ff2 = (const float*)d_in[14];
  float* out = (float*)d_out;

  char* ws = (char*)d_ws;
  size_t off = 0;
  auto alloc = [&](size_t bytes) -> void* {
    void* p = ws + off;
    off = (off + bytes + 255) & ~(size_t)255;
    return p;
  };
  u16*   wqkv_bf = (u16*)alloc((size_t)4 * 1024 * 3072 * 2);
  u16*   wr_bf   = (u16*)alloc((size_t)1024 * 1024 * 2);
  u16*   wo_bf   = (u16*)alloc((size_t)4 * 1024 * 1024 * 2);
  u16*   wff1_bf = (u16*)alloc((size_t)4 * 1024 * 4096 * 2);
  u16*   wff2_bf = (u16*)alloc((size_t)4 * 4096 * 1024 * 2);
  u16*   p_bf    = (u16*)alloc((size_t)2048 * 1024 * 2);
  u16*   r_bf    = (u16*)alloc((size_t)2048 * 1024 * 2);
  u16*   xmem_bf = (u16*)alloc((size_t)2 * 2048 * 1024 * 2);
  u16*   heads_bf= (u16*)alloc((size_t)2 * 2048 * 3072 * 2);
  u16*   qrw_bf  = (u16*)alloc((size_t)2 * 1024 * 1024 * 2);
  u16*   qrr_bf  = (u16*)alloc((size_t)2 * 1024 * 1024 * 2);
  u16*   ao_bf   = (u16*)alloc((size_t)2 * 1024 * 1024 * 2);
  u16*   ff_bf   = (u16*)alloc((size_t)2 * 1024 * 4096 * 2);
  float* gout    = (float*)alloc((size_t)2 * 1024 * 1024 * 4);
  float* xcur    = (float*)alloc((size_t)2 * 1024 * 1024 * 4);
  u16*   xcur_bf = (u16*)alloc((size_t)2 * 1024 * 1024 * 2);

  auto gemm = [&](const u16* A, const u16* Bm, const float* bias, float* Cf, u16* Cb,
                  int M, int N, int K, int flags) {
    dim3 g(N / 128, M / 128);
    switch (flags) {
      case 0:
        gemm_bf16_kernel<0><<<g, 256, 0, stream>>>(A, Bm, bias, Cf, Cb, M, N, K);
        break;
      case FLAG_OUT_BF16:
        gemm_bf16_kernel<FLAG_OUT_BF16><<<g, 256, 0, stream>>>(A, Bm, bias, Cf, Cb, M, N, K);
        break;
      case FLAG_BIAS:
        gemm_bf16_kernel<FLAG_BIAS><<<g, 256, 0, stream>>>(A, Bm, bias, Cf, Cb, M, N, K);
        break;
      default:
        gemm_bf16_kernel<FLAG_OUT_BF16 | FLAG_BIAS | FLAG_RELU>
            <<<g, 256, 0, stream>>>(A, Bm, bias, Cf, Cb, M, N, K);
        break;
    }
  };

  // one-time conversions + positional embedding + r projection + x init
  cvt_kernel<<<(4 * 1024 * 3072) / 1024, 256, 0, stream>>>(W_qkv, wqkv_bf);
  cvt_kernel<<<(1024 * 1024) / 1024,     256, 0, stream>>>(W_r,   wr_bf);
  cvt_kernel<<<(4 * 1024 * 1024) / 1024, 256, 0, stream>>>(W_o,   wo_bf);
  cvt_kernel<<<(4 * 1024 * 4096) / 1024, 256, 0, stream>>>(W_ff1, wff1_bf);
  cvt_kernel<<<(4 * 4096 * 1024) / 1024, 256, 0, stream>>>(W_ff2, wff2_bf);
  pos_kernel<<<2048, 256, 0, stream>>>(p_bf);
  gemm(p_bf, wr_bf, nullptr, nullptr, r_bf, 2048, 1024, 1024, FLAG_OUT_BF16);
  initx_kernel<<<(2 * 1024 * 1024) / 256, 256, 0, stream>>>(x_in, xcur, xcur_bf);

  for (int l = 0; l < 4; ++l) {
    concat_kernel<<<(2 * 2048 * 1024) / 256, 256, 0, stream>>>(
        mem + (size_t)l * 2 * 1024 * 1024, xcur, xmem_bf);
    gemm(xmem_bf, wqkv_bf + (size_t)l * 1024 * 3072, nullptr, nullptr, heads_bf,
         4096, 3072, 1024, FLAG_OUT_BF16);
    qbias_kernel<<<(2 * 1024 * 1024) / 256, 256, 0, stream>>>(heads_bf, rwb, rrb,
                                                              qrw_bf, qrr_bf);
    attn_kernel<<<dim3(16, 16, 2), 128, 0, stream>>>(heads_bf, qrw_bf, qrr_bf,
                                                     r_bf, ao_bf);
    gemm(ao_bf, wo_bf + (size_t)l * 1024 * 1024, nullptr, gout, nullptr,
         2048, 1024, 1024, 0);
    add_ln_kernel<<<2048, 256, 0, stream>>>(gout, xcur, xcur_bf,
                                            ln1g + l * 1024, ln1b + l * 1024);
    gemm(xcur_bf, wff1_bf + (size_t)l * 1024 * 4096, b_ff1 + l * 4096, nullptr,
         ff_bf, 2048, 4096, 1024, FLAG_OUT_BF16 | FLAG_BIAS | FLAG_RELU);
    gemm(ff_bf, wff2_bf + (size_t)l * 4096 * 1024, b_ff2 + l * 1024, gout, nullptr,
         2048, 1024, 4096, FLAG_BIAS);
    add_ln_kernel<<<2048, 256, 0, stream>>>(gout, xcur, xcur_bf,
                                            ln2g + l * 1024, ln2b + l * 1024);
  }
  (void)hipMemcpyAsync(out, xcur, (size_t)2 * 1024 * 1024 * 4,
                       hipMemcpyDeviceToDevice, stream);
}
